// NeurTWs_56822417326739
// MI455X (gfx1250) — matrix operand order repaired and checked by reading, not compiled, not run on hardware
//
#include <hip/hip_runtime.h>

typedef __attribute__((ext_vector_type(16))) _Float16 v16h;
typedef __attribute__((ext_vector_type(8)))  _Float16 v8h;
typedef __attribute__((ext_vector_type(4)))  _Float16 v4h;
typedef __attribute__((ext_vector_type(8)))  float    v8f;
typedef __attribute__((ext_vector_type(4)))  float    v4f;

#define LP_    4
#define ENC_   32
#define FEAT_  64
#define XDIM_  96      // ENC_ + FEAT_
#define D3_    64
#define XSTR_  104     // padded LDS row stride (halves); 104*2 = 208 B, 16B-aligned rows
#define RPW_   16      // rows per wave-tile (one WMMA M-tile)
#define WPB_   8       // waves per 256-thread block
#define TPW_   8       // tiles per wave (amortizes weight staging)
#define RPB_   (WPB_ * TPW_ * RPW_)   // 1024 rows per block

static __device__ __forceinline__ v16h cat16(v8h lo, v8h hi) {
  return __builtin_shufflevector(lo, hi, 0,1,2,3,4,5,6,7,8,9,10,11,12,13,14,15);
}

__global__ __launch_bounds__(256) void neurtw_fused(
    const float* __restrict__ pos_table, const float* __restrict__ node_feat,
    const float* __restrict__ W1,  const float* __restrict__ b1,
    const float* __restrict__ W2,  const float* __restrict__ b2,
    const float* __restrict__ Wm1, const float* __restrict__ bm1,
    const float* __restrict__ Wm2, const float* __restrict__ bm2,
    const int* __restrict__ key_idx, const int* __restrict__ node_idx,
    float* __restrict__ out)
{
  // LDS: only the big reused weight (Wm1^T, f16) and the per-wave x staging buffer.
  __shared__ __attribute__((aligned(16))) _Float16 s_wm1t[D3_*XDIM_];    // [n][k]
  __shared__ __attribute__((aligned(16))) _Float16 s_x[WPB_*RPW_*XSTR_]; // per-wave [pe|feat]

  const int tid = threadIdx.x;

  // ---- stage Wm1^T in LDS as f16 (once per 1024 rows)
  for (int i = tid; i < D3_*XDIM_; i += 256) {
    int n = i / XDIM_, k = i - n*XDIM_;
    s_wm1t[n*XDIM_ + k] = (_Float16)Wm1[k*D3_ + n];
  }
  __syncthreads();

  const int wave = tid >> 5;
  const int lane = tid & 31;
  const int m  = lane & 15;   // M-row (A) / N-col (B,C,D) index
  const int kh = lane >> 4;   // lane half
  _Float16* xb = &s_x[wave * RPW_ * XSTR_];

  // ---- per-wave invariants built straight from global (L2-resident, once per 128 rows)
  // W1 B fragments: K=4 zero-padded to 32. Lane half 0 carries K=0..3 in elements 0..3.
  v16h bW1[2];
#pragma unroll
  for (int t = 0; t < 2; ++t) {
    const int n = t*16 + m;
    v16h z;
#pragma unroll
    for (int i = 0; i < 16; ++i) z[i] = (_Float16)0.f;
    if (kh == 0) {
      z[0] = (_Float16)W1[0*ENC_ + n];
      z[1] = (_Float16)W1[1*ENC_ + n];
      z[2] = (_Float16)W1[2*ENC_ + n];
      z[3] = (_Float16)W1[3*ENC_ + n];
    }
    bW1[t] = z;
  }
  // W2 B fragments: col n, K = kh*8..+7 then 16+kh*8..+7 (strided column reads from global, once)
  v16h bW2[2];
#pragma unroll
  for (int t = 0; t < 2; ++t) {
    const int n = t*16 + m;
    v16h z;
#pragma unroll
    for (int i = 0; i < 16; ++i) z[i] = (_Float16)W2[(i < 8 ? kh*8 + i : 8 + kh*8 + i)*ENC_ + n];
    bW2[t] = z;
  }
  const float b1_0 = b1[m],     b1_1 = b1[16 + m];
  const float b2_0 = b2[m],     b2_1 = b2[16 + m];
  float bm1v[4], wm2v[4];
#pragma unroll
  for (int t = 0; t < 4; ++t) { bm1v[t] = bm1[t*16 + m]; wm2v[t] = Wm2[t*16 + m]; }
  const float bb = bm2[0];
  const v8f vzero = {0.f,0.f,0.f,0.f,0.f,0.f,0.f,0.f};

  const int waveRow0 = blockIdx.x * RPB_ + wave * (TPW_ * RPW_);

#pragma unroll 1
  for (int it = 0; it < TPW_; ++it) {
    const int row0 = waveRow0 + it * RPW_;

    // Opaque zero: loop-variant as far as the compiler can prove, so the Wm1 B-fragment
    // LDS loads below CANNOT be hoisted out of the loop (keeps VGPR pressure < 256).
    int oz = 0;
    asm volatile("" : "+v"(oz));

    // ---- enc gather (16B per row); A fragment built directly in registers (K=4 padded)
    const int key = key_idx[row0 + m];
    const v4f e4 = *(const v4f*)(pos_table + key * LP_);

    // ---- feat gather: 2 lanes per row, 32 floats each -> f16 into x cols 32..95
    {
      const int r = lane >> 1, hf = lane & 1;
      const int nid = node_idx[row0 + r];
      const float* fp = node_feat + nid * FEAT_ + hf * 32;
      _Float16* dst = &xb[r*XSTR_ + ENC_ + hf*32];
#pragma unroll
      for (int j = 0; j < 8; ++j) {
        v4f f = *(const v4f*)(fp + 4*j);                     // global_load_b128
        v4h h = { (_Float16)f.x, (_Float16)f.y, (_Float16)f.z, (_Float16)f.w };
        *(v4h*)(dst + 4*j) = h;                              // ds_store_b64
      }
    }

    // ---- layer 1 as WMMA: h1 = relu(enc @ W1 + b1); two 16x16x32 WMMAs, K zero-padded
    {
      v16h Aenc;
#pragma unroll
      for (int i = 0; i < 16; ++i) Aenc[i] = (_Float16)0.f;
      if (kh == 0) {
        Aenc[0] = (_Float16)e4.x; Aenc[1] = (_Float16)e4.y;
        Aenc[2] = (_Float16)e4.z; Aenc[3] = (_Float16)e4.w;
      }
      v8f c0 = __builtin_amdgcn_wmma_f32_16x16x32_f16(false, Aenc, false, bW1[0],
                                                      (short)0, vzero, false, false);
      v8f c1 = __builtin_amdgcn_wmma_f32_16x16x32_f16(false, Aenc, false, bW1[1],
                                                      (short)0, vzero, false, false);
      // C layout: lane holds col n, rows kh*8+v. bias + relu, write f16 to x cols 0..31
#pragma unroll
      for (int v = 0; v < 8; ++v) {
        float h0 = c0[v] + b1_0; h0 = h0 > 0.f ? h0 : 0.f;
        float h1 = c1[v] + b1_1; h1 = h1 > 0.f ? h1 : 0.f;
        xb[(kh*8 + v)*XSTR_ + m]      = (_Float16)h0;
        xb[(kh*8 + v)*XSTR_ + 16 + m] = (_Float16)h1;
      }
    }

    // ---- layer 2: pe = h1 @ W2 + b2 -> two WMMAs; write back into x cols 0..31
    {
      v8h alo = *(const v8h*)&xb[m*XSTR_ + kh*8];
      v8h ahi = *(const v8h*)&xb[m*XSTR_ + 16 + kh*8];
      v16h A = cat16(alo, ahi);
      v8f pe0 = __builtin_amdgcn_wmma_f32_16x16x32_f16(false, A, false, bW2[0],
                                                       (short)0, vzero, false, false);
      v8f pe1 = __builtin_amdgcn_wmma_f32_16x16x32_f16(false, A, false, bW2[1],
                                                       (short)0, vzero, false, false);
#pragma unroll
      for (int v = 0; v < 8; ++v) {
        xb[(kh*8 + v)*XSTR_ + m]      = (_Float16)(pe0[v] + b2_0);
        xb[(kh*8 + v)*XSTR_ + 16 + m] = (_Float16)(pe1[v] + b2_1);
      }
    }

    // ---- merge fc1: h2 = x(16x96) @ Wm1(96x64); 3 K-chunks x 4 N-tiles = 12 WMMAs.
    // B fragments streamed from LDS each iteration (oz keeps them un-hoistable).
    v8f acc[4];
#pragma unroll
    for (int t = 0; t < 4; ++t) acc[t] = vzero;
#pragma unroll
    for (int kc = 0; kc < 3; ++kc) {
      const int K0 = kc * 32;
      v8h alo = *(const v8h*)&xb[m*XSTR_ + K0 + kh*8];
      v8h ahi = *(const v8h*)&xb[m*XSTR_ + K0 + 16 + kh*8];
      v16h A = cat16(alo, ahi);
#pragma unroll
      for (int t = 0; t < 4; ++t) {
        const _Float16* wr = &s_wm1t[(t*16 + m)*XDIM_ + K0 + kh*8 + oz];
        v16h Bv = cat16(*(const v8h*)wr, *(const v8h*)(wr + 16)); // 2x ds_load_b128
        acc[t] = __builtin_amdgcn_wmma_f32_16x16x32_f16(false, A, false, Bv,
                                                        (short)0, acc[t], false, false);
      }
    }

    // ---- fc2 (64 -> 1): relu + matvec in VALU on C-layout regs, xor-reduce over N lanes
    float s[8];
#pragma unroll
    for (int v = 0; v < 8; ++v) s[v] = 0.f;
#pragma unroll
    for (int t = 0; t < 4; ++t) {
#pragma unroll
      for (int v = 0; v < 8; ++v) {
        float h = acc[t][v] + bm1v[t];
        h = h > 0.f ? h : 0.f;
        s[v] = fmaf(h, wm2v[t], s[v]);
      }
    }
#pragma unroll
    for (int mask = 1; mask <= 8; mask <<= 1) {
#pragma unroll
      for (int v = 0; v < 8; ++v) s[v] += __shfl_xor(s[v], mask, 32);
    }
    if (m == 0) {          // lanes 0 and 16 hold rows kh*8 .. kh*8+7
#pragma unroll
      for (int v = 0; v < 8; ++v) out[row0 + kh*8 + v] = s[v] + bb;
    }
  }
}

extern "C" void kernel_launch(void* const* d_in, const int* in_sizes, int n_in,
                              void* d_out, int out_size, void* d_ws, size_t ws_size,
                              hipStream_t stream) {
  const float* pos_table = (const float*)d_in[0];
  const float* node_feat = (const float*)d_in[1];
  const float* W1  = (const float*)d_in[2];
  const float* b1  = (const float*)d_in[3];
  const float* W2  = (const float*)d_in[4];
  const float* b2  = (const float*)d_in[5];
  const float* Wm1 = (const float*)d_in[6];
  const float* bm1 = (const float*)d_in[7];
  const float* Wm2 = (const float*)d_in[8];
  const float* bm2 = (const float*)d_in[9];
  const int* key_idx  = (const int*)d_in[10];
  const int* node_idx = (const int*)d_in[11];
  float* out = (float*)d_out;

  const int rows = in_sizes[10];          // B*M = 1,048,576
  const int blocks = rows / RPB_;         // 1024 blocks of 256 threads (8 waves x 8 tiles)
  neurtw_fused<<<blocks, 256, 0, stream>>>(pos_table, node_feat, W1, b1, W2, b2,
                                           Wm1, bm1, Wm2, bm2, key_idx, node_idx, out);
}
